// SimpleTacotron_80238579024236
// MI455X (gfx1250) — matrix operand order repaired and checked by reading, not compiled, not run on hardware
//
#include <hip/hip_runtime.h>

// ---------------------------------------------------------------------------
// SimpleTacotron on gfx1250: WMMA f16 (f32 accum), cluster-of-16-WG decoder
// ---------------------------------------------------------------------------

typedef __attribute__((ext_vector_type(16))) _Float16 v16h;
typedef __attribute__((ext_vector_type(8)))  _Float16 v8h;
typedef __attribute__((ext_vector_type(8)))  float    v8f;

#define WMMA_F16(a, b, c) \
  __builtin_amdgcn_wmma_f32_16x16x32_f16(false, (a), false, (b), (short)0, (c), false, false)

#define NWG 16  // decoder workgroups

// A fragment (16x32 f16, M x K): row-major source, ld in halves.
// Lane L holds row M=L&15; lane-half g selects the documented K striping
// -> two contiguous 16B loads per lane.
__device__ __forceinline__ v16h load_a_frag(const _Float16* X, int ld, int row, int k0, int g) {
  const _Float16* p = X + (long)row * ld + k0 + 8 * g;
  v8h lo = *(const v8h*)(p);
  v8h hi = *(const v8h*)(p + 16);
  v16h a;
#pragma unroll
  for (int i = 0; i < 8; ++i) { a[i] = lo[i]; a[i + 8] = hi[i]; }
  return a;
}

// B fragment from pre-packed weights: tile = 512 halves, lane-major contiguous.
__device__ __forceinline__ v16h load_b_frag(const _Float16* P, long tile, int lane) {
  return *(const v16h*)(P + tile * 512 + (long)lane * 16);
}

__device__ __forceinline__ float sigm(float x) { return 1.f / (1.f + __expf(-x)); }

// Device-scope sense-reversing barrier across the decoder's resident WGs.
__device__ __forceinline__ void grid_barrier(unsigned* cnt, unsigned* gen, unsigned nwg) {
  __syncthreads();
  if (threadIdx.x == 0) {
    unsigned g = __hip_atomic_load(gen, __ATOMIC_RELAXED, __HIP_MEMORY_SCOPE_AGENT);
    if (__hip_atomic_fetch_add(cnt, 1u, __ATOMIC_ACQ_REL, __HIP_MEMORY_SCOPE_AGENT) == nwg - 1) {
      __hip_atomic_store(cnt, 0u, __ATOMIC_RELAXED, __HIP_MEMORY_SCOPE_AGENT);
      __hip_atomic_fetch_add(gen, 1u, __ATOMIC_ACQ_REL, __HIP_MEMORY_SCOPE_AGENT);
    } else {
      while (__hip_atomic_load(gen, __ATOMIC_ACQUIRE, __HIP_MEMORY_SCOPE_AGENT) == g)
        __builtin_amdgcn_s_sleep(1);
    }
  }
  __syncthreads();
}

// ---------------------------------------------------------------------------
// Weight packer: packed[(nt*ktiles+kt)*512 + lane*16 + i] =
//   W[nt*16 + (lane&15)][koff + kt*32 + 8*(lane>>4) + (i<8 ? i : 8+i)]   (0 outside valid)
// ---------------------------------------------------------------------------
__global__ void pack_w(const float* __restrict__ W, int ldW, int koff,
                       int nvalid, int kvalid, int ntiles, int ktiles,
                       _Float16* __restrict__ out) {
  long total = (long)ntiles * ktiles * 512;
  for (long e = (long)blockIdx.x * blockDim.x + threadIdx.x; e < total;
       e += (long)gridDim.x * blockDim.x) {
    long tile = e >> 9;
    int r = (int)(e & 511);
    int lane = r >> 4, i = r & 15;
    int nt = (int)(tile / ktiles), kt = (int)(tile % ktiles);
    int n = nt * 16 + (lane & 15);
    int k = kt * 32 + 8 * (lane >> 4) + (i < 8 ? i : 8 + i);
    float v = 0.f;
    if (n < nvalid && k < kvalid) v = W[(long)n * ldW + koff + k];
    out[e] = (_Float16)v;
  }
}

__global__ void embed_k(const int* __restrict__ text, const float* __restrict__ emb,
                        _Float16* __restrict__ xemb) {
  long total = 200L * 32 * 256;
  for (long e = (long)blockIdx.x * blockDim.x + threadIdx.x; e < total;
       e += (long)gridDim.x * blockDim.x) {
    int k = (int)(e & 255);
    long bt = e >> 8;
    int b = (int)(bt & 31);
    int t = (int)(bt >> 5);
    int tok = text[b * 200 + t];
    xemb[e] = (_Float16)emb[(long)tok * 256 + k];
  }
}

__global__ void decin_k(const float* __restrict__ mel, _Float16* __restrict__ decin) {
  long total = 400L * 32 * 80;
  for (long e = (long)blockIdx.x * blockDim.x + threadIdx.x; e < total;
       e += (long)gridDim.x * blockDim.x) {
    int o = (int)(e % 80);
    long tb = e / 80;
    int b = (int)(tb % 32);
    int t = (int)(tb / 32);
    float v = (t == 0) ? 0.f : mel[((long)b * 400 + (t - 1)) * 80 + o];
    decin[e] = (_Float16)v;
  }
}

__global__ void zero_bar(unsigned* bar) {
  if (threadIdx.x < 2) bar[threadIdx.x] = 0u;
}

// ---------------------------------------------------------------------------
// Encoder LSTM: one persistent block per direction (grid=2), 256 threads.
// Dynamic LDS: gates f32 [32][1024] | c f32 [32][256] | h f16 [32][256] = 176 KB
// ---------------------------------------------------------------------------
__global__ void __launch_bounds__(256) enc_lstm(
    const _Float16* __restrict__ xemb,
    const _Float16* __restrict__ Pf_ih, const _Float16* __restrict__ Pf_hh,
    const float* __restrict__ bf,
    const _Float16* __restrict__ Pb_ih, const _Float16* __restrict__ Pb_hh,
    const float* __restrict__ bb,
    float* __restrict__ hf, float* __restrict__ hb) {
  extern __shared__ char smem[];
  float* gates = (float*)smem;
  float* cbuf = (float*)(smem + 131072);
  _Float16* h16 = (_Float16*)(smem + 131072 + 32768);
  int tid = threadIdx.x, lane = tid & 31, wv = tid >> 5;
  int g = lane >> 4, col = lane & 15;
  int dir = blockIdx.x;
  const _Float16* Pih = dir ? Pb_ih : Pf_ih;
  const _Float16* Phh = dir ? Pb_hh : Pf_hh;
  const float* bias = dir ? bb : bf;
  float* hout = dir ? hb : hf;

  for (int e = tid; e < 32 * 256; e += 256) { cbuf[e] = 0.f; h16[e] = (_Float16)0.f; }
  __syncthreads();

  for (int step = 0; step < 200; ++step) {
    int t = dir ? (199 - step) : step;
    const _Float16* xt = xemb + (long)t * 32 * 256;
    for (int tile = wv; tile < 128; tile += 8) {
      int m = tile >> 6, n = tile & 63;
      int row = m * 16 + col;
      v8f acc = {};
#pragma unroll
      for (int kt = 0; kt < 8; ++kt) {
        v16h a = load_a_frag(xt, 256, row, kt * 32, g);
        v16h b = load_b_frag(Pih, (long)n * 8 + kt, lane);
        acc = WMMA_F16(a, b, acc);
      }
#pragma unroll
      for (int kt = 0; kt < 8; ++kt) {
        v16h a = load_a_frag(h16, 256, row, kt * 32, g);
        v16h b = load_b_frag(Phh, (long)n * 8 + kt, lane);
        acc = WMMA_F16(a, b, acc);
      }
#pragma unroll
      for (int j = 0; j < 8; ++j)
        gates[(m * 16 + 8 * g + j) * 1024 + n * 16 + col] = acc[j];
    }
    __syncthreads();
    for (int e = tid; e < 32 * 256; e += 256) {
      int b_ = e >> 8, j = e & 255;
      float gi = gates[b_ * 1024 + j] + bias[j];
      float gf = gates[b_ * 1024 + 256 + j] + bias[256 + j];
      float gg = gates[b_ * 1024 + 512 + j] + bias[512 + j];
      float go = gates[b_ * 1024 + 768 + j] + bias[768 + j];
      float c = sigm(gf) * cbuf[e] + sigm(gi) * tanhf(gg);
      float h = sigm(go) * tanhf(c);
      cbuf[e] = c;
      h16[e] = (_Float16)h;
      hout[(long)t * 32 * 256 + e] = h;
    }
    __syncthreads();
  }
}

__global__ void assemble_enc(const float* __restrict__ hf, const float* __restrict__ hb,
                             float* __restrict__ enc32, _Float16* __restrict__ enc16) {
  long total = 6400L * 512;
  for (long e = (long)blockIdx.x * blockDim.x + threadIdx.x; e < total;
       e += (long)gridDim.x * blockDim.x) {
    int ec = (int)(e & 511);
    long r = e >> 9;
    int s = (int)(r % 200);
    int b = (int)(r / 200);
    float v = (ec < 256) ? hf[((long)s * 32 + b) * 256 + ec]
                         : hb[((long)s * 32 + b) * 256 + (ec - 256)];
    enc32[e] = v;
    enc16[e] = (_Float16)v;
  }
}

// proj_enc = enc_out @ Ws_enc^T + bs  (M=6400, N=1024, K=512); 1 tile per wave.
__global__ void __launch_bounds__(256) proj_gemm(const _Float16* __restrict__ enc16,
                                                 const _Float16* __restrict__ Pws,
                                                 const float* __restrict__ bs,
                                                 float* __restrict__ proj) {
  int tid = threadIdx.x, lane = tid & 31, wv = tid >> 5;
  int g = lane >> 4, col = lane & 15;
  int tile = blockIdx.x * 8 + wv;
  int mt = tile >> 6, nt = tile & 63;
  int row = mt * 16 + col;
  v8f acc = {};
#pragma unroll
  for (int kt = 0; kt < 16; ++kt) {
    v16h a = load_a_frag(enc16, 512, row, kt * 32, g);
    v16h b = load_b_frag(Pws, (long)nt * 16 + kt, lane);
    acc = WMMA_F16(a, b, acc);
  }
  float bv = bs[nt * 16 + col];
#pragma unroll
  for (int j = 0; j < 8; ++j)
    proj[(long)(mt * 16 + 8 * g + j) * 1024 + nt * 16 + col] = acc[j] + bv;
}

// ---------------------------------------------------------------------------
// Decoder: 16 cooperating persistent WGs, 400 sequential steps.
// WG w owns batch rows {2w, 2w+1} for attention (scores/softmax stay in
// LDS) and gate group (m,n) = wg*8+wv (one per wave); c state is WG-local LDS.
// Shared per step via L2: x_in slices (gxin) and new h (gh double-buffered).
// Static LDS per WG: 32KB sdec + 8KB c + 1.6KB scores = 42.5 KB.
// ---------------------------------------------------------------------------
__global__ void __launch_bounds__(256) decoder_k(
    const float* __restrict__ proj, const float* __restrict__ enc32,
    const _Float16* __restrict__ decin,
    const _Float16* __restrict__ Pwsd, const _Float16* __restrict__ Pwih,
    const _Float16* __restrict__ Pwhh, const _Float16* __restrict__ Pwout,
    const _Float16* __restrict__ Pwstop,
    const float* __restrict__ vvec, const float* __restrict__ dec_b,
    const float* __restrict__ bout, const float* __restrict__ bstop,
    _Float16* __restrict__ gxin, _Float16* __restrict__ gh0,
    _Float16* __restrict__ gh1, unsigned* bar,
    float* __restrict__ out_mel, float* __restrict__ out_stop,
    float* __restrict__ out_align) {
  __shared__ _Float16 s_sdec[16 * 1024];  // sdec m-tile holding our 2 rows
  __shared__ float s_c[8 * 256];          // c slice for this WG's 8 gate groups
  __shared__ float s_sc[2 * 200];         // scores/attention weights, 2 rows
  int tid = threadIdx.x, lane = tid & 31, wv = tid >> 5;
  int g = lane >> 4, col = lane & 15;
  int wg = blockIdx.x;
  int b0 = wg * 2;       // first owned batch row
  int mt = b0 >> 4;      // m-tile containing rows b0, b0+1

  // init: c slice, h double-buffers, x_in (incl. zero K-padding)
  for (int e = tid; e < 8 * 256; e += 256) s_c[e] = 0.f;
  for (int e = wg * 256 + tid; e < 32 * 1024; e += NWG * 256) {
    gh0[e] = (_Float16)0.f;
    gh1[e] = (_Float16)0.f;
  }
  if (wg == 0)
    for (int e = tid; e < 32 * 608; e += 256) gxin[e] = (_Float16)0.f;
  grid_barrier(bar, bar + 1, NWG);

  for (int t = 0; t < 400; ++t) {
    const _Float16* hr = (t & 1) ? gh1 : gh0;  // current h
    _Float16* hw = (t & 1) ? gh0 : gh1;        // next h
    // --- phase 1: sdec(m-tile) = h @ Ws_dec^T (16 rows x 1024), WG-local
    for (int n = wv; n < 64; n += 8) {
      int row = mt * 16 + col;
      v8f acc = {};
      for (int kt = 0; kt < 32; ++kt) {
        if (kt + 4 < 32)
          __builtin_prefetch(Pwsd + ((long)n * 32 + kt + 4) * 512 + (long)lane * 16, 0, 0);
        v16h a = load_a_frag(hr, 1024, row, kt * 32, g);
        v16h b = load_b_frag(Pwsd, (long)n * 32 + kt, lane);
        acc = WMMA_F16(a, b, acc);
      }
#pragma unroll
      for (int j = 0; j < 8; ++j)
        s_sdec[(8 * g + j) * 1024 + n * 16 + col] = (_Float16)acc[j];
    }
    __syncthreads();
    // --- phase 2: scores[b][s] = sum_d v[d]*tanh(proj[b][s][d]+sdec[b][d])
    for (int p = tid; p < 400; p += 256) {
      int lr = (p >= 200) ? 1 : 0;
      int b_ = b0 + lr, s = p - lr * 200;
      const float* pe = proj + ((long)b_ * 200 + s) * 1024;
      const _Float16* sd = s_sdec + (b_ & 15) * 1024;
      float sum = 0.f;
      for (int d = 0; d < 1024; d += 4) {
        sum += vvec[d + 0] * tanhf(pe[d + 0] + (float)sd[d + 0]);
        sum += vvec[d + 1] * tanhf(pe[d + 1] + (float)sd[d + 1]);
        sum += vvec[d + 2] * tanhf(pe[d + 2] + (float)sd[d + 2]);
        sum += vvec[d + 3] * tanhf(pe[d + 3] + (float)sd[d + 3]);
      }
      s_sc[lr * 200 + s] = sum;
    }
    __syncthreads();
    // --- phase 3: softmax over s for our 2 rows; emit alignments
    if (tid < 2) {
      int b_ = b0 + tid;
      float mx = -1e30f;
      for (int s = 0; s < 200; ++s) mx = fmaxf(mx, s_sc[tid * 200 + s]);
      float sum = 0.f;
      for (int s = 0; s < 200; ++s) {
        float e2 = __expf(s_sc[tid * 200 + s] - mx);
        s_sc[tid * 200 + s] = e2;
        sum += e2;
      }
      float inv = 1.f / sum;
      for (int s = 0; s < 200; ++s) {
        float w = s_sc[tid * 200 + s] * inv;
        s_sc[tid * 200 + s] = w;
        out_align[((long)b_ * 400 + t) * 200 + s] = w;
      }
    }
    __syncthreads();
    // --- phase 4: ctx for our 2 rows -> gxin; WG0 also fills x_t part
    for (int q = tid; q < 1024; q += 256) {
      int lr = q >> 9, e2 = q & 511;
      int b_ = b0 + lr;
      const float* eo = enc32 + (long)b_ * 200 * 512 + e2;
      float sum = 0.f;
      for (int s = 0; s < 200; ++s) sum += s_sc[lr * 200 + s] * eo[(long)s * 512];
      gxin[b_ * 608 + 80 + e2] = (_Float16)sum;
    }
    if (wg == 0)
      for (int q = tid; q < 2560; q += 256)
        gxin[(q / 80) * 608 + (q % 80)] = decin[(long)t * 2560 + q];
    grid_barrier(bar, bar + 1, NWG);
    // --- phase 5: one gate group (m,n) per wave; i/f/g/o in registers
    {
      int gid = wg * 8 + wv;
      int m = gid >> 6, n = gid & 63;
      int row = m * 16 + col;
      v8f acc0 = {}, acc1 = {}, acc2 = {}, acc3 = {};
      for (int kt = 0; kt < 19; ++kt) {
        v16h a = load_a_frag(gxin, 608, row, kt * 32, g);
        acc0 = WMMA_F16(a, load_b_frag(Pwih, (long)(0 * 64 + n) * 19 + kt, lane), acc0);
        acc1 = WMMA_F16(a, load_b_frag(Pwih, (long)(1 * 64 + n) * 19 + kt, lane), acc1);
        acc2 = WMMA_F16(a, load_b_frag(Pwih, (long)(2 * 64 + n) * 19 + kt, lane), acc2);
        acc3 = WMMA_F16(a, load_b_frag(Pwih, (long)(3 * 64 + n) * 19 + kt, lane), acc3);
      }
      for (int kt = 0; kt < 32; ++kt) {
        v16h a = load_a_frag(hr, 1024, row, kt * 32, g);
        acc0 = WMMA_F16(a, load_b_frag(Pwhh, (long)(0 * 64 + n) * 32 + kt, lane), acc0);
        acc1 = WMMA_F16(a, load_b_frag(Pwhh, (long)(1 * 64 + n) * 32 + kt, lane), acc1);
        acc2 = WMMA_F16(a, load_b_frag(Pwhh, (long)(2 * 64 + n) * 32 + kt, lane), acc2);
        acc3 = WMMA_F16(a, load_b_frag(Pwhh, (long)(3 * 64 + n) * 32 + kt, lane), acc3);
      }
      float bi = dec_b[n * 16 + col];
      float bf2 = dec_b[1024 + n * 16 + col];
      float bg = dec_b[2048 + n * 16 + col];
      float bo = dec_b[3072 + n * 16 + col];
#pragma unroll
      for (int j = 0; j < 8; ++j) {
        int lidx = wv * 256 + (8 * g + j) * 16 + col;           // local c slot
        int hidx = (m * 16 + 8 * g + j) * 1024 + n * 16 + col;  // global h slot
        float c = sigm(acc1[j] + bf2) * s_c[lidx] + sigm(acc0[j] + bi) * tanhf(acc2[j] + bg);
        float h = sigm(acc3[j] + bo) * tanhf(c);
        s_c[lidx] = c;
        hw[hidx] = (_Float16)h;
      }
    }
    grid_barrier(bar, bar + 1, NWG);
    // --- phase 6 (WG0): mel = h@Wout^T + bout ; stop = sigmoid(h@Wstop^T+bstop)
    if (wg == 0) {
      for (int tile = wv; tile < 12; tile += 8) {
        int m = (tile >= 6) ? 1 : 0;
        int nt = tile % 6;
        int row = m * 16 + col;
        v8f acc = {};
        const _Float16* PB = (nt < 5) ? Pwout : Pwstop;
        long nbase = (nt < 5) ? (long)nt * 32 : 0;
        for (int kt = 0; kt < 32; ++kt) {
          v16h a = load_a_frag(hw, 1024, row, kt * 32, g);
          v16h b = load_b_frag(PB, nbase + kt, lane);
          acc = WMMA_F16(a, b, acc);
        }
        if (nt < 5) {
          float bv = bout[nt * 16 + col];
#pragma unroll
          for (int j = 0; j < 8; ++j) {
            int b_ = m * 16 + 8 * g + j;
            out_mel[((long)b_ * 400 + t) * 80 + nt * 16 + col] = acc[j] + bv;
          }
        } else if (col == 0) {
          float bv = bstop[0];
#pragma unroll
          for (int j = 0; j < 8; ++j) {
            int b_ = m * 16 + 8 * g + j;
            out_stop[(long)b_ * 400 + t] = sigm(acc[j] + bv);
          }
        }
      }
    }
  }
}

// ---------------------------------------------------------------------------
extern "C" void kernel_launch(void* const* d_in, const int* in_sizes, int n_in,
                              void* d_out, int out_size, void* d_ws, size_t ws_size,
                              hipStream_t stream) {
  const int* text = (const int*)d_in[0];
  const float* mel = (const float*)d_in[1];
  const float* emb = (const float*)d_in[2];
  const float* efWih = (const float*)d_in[3];
  const float* efWhh = (const float*)d_in[4];
  const float* efb = (const float*)d_in[5];
  const float* ebWih = (const float*)d_in[6];
  const float* ebWhh = (const float*)d_in[7];
  const float* ebb = (const float*)d_in[8];
  const float* Ws = (const float*)d_in[9];
  const float* bs = (const float*)d_in[10];
  const float* v = (const float*)d_in[11];
  const float* dWih = (const float*)d_in[12];
  const float* dWhh = (const float*)d_in[13];
  const float* db = (const float*)d_in[14];
  const float* Wout = (const float*)d_in[15];
  const float* bout = (const float*)d_in[16];
  const float* Wstop = (const float*)d_in[17];
  const float* bstop = (const float*)d_in[18];

  char* ws = (char*)d_ws;
  size_t off = 0;
  auto alloc = [&](size_t bytes) {
    size_t o = off;
    off += (bytes + 255) & ~(size_t)255;
    return o;
  };
  // packed f16 weights
  _Float16* P_efWih = (_Float16*)(ws + alloc(64L * 8 * 512 * 2));
  _Float16* P_efWhh = (_Float16*)(ws + alloc(64L * 8 * 512 * 2));
  _Float16* P_ebWih = (_Float16*)(ws + alloc(64L * 8 * 512 * 2));
  _Float16* P_ebWhh = (_Float16*)(ws + alloc(64L * 8 * 512 * 2));
  _Float16* P_Wse = (_Float16*)(ws + alloc(64L * 16 * 512 * 2));
  _Float16* P_Wsd = (_Float16*)(ws + alloc(64L * 32 * 512 * 2));
  _Float16* P_dWih = (_Float16*)(ws + alloc(256L * 19 * 512 * 2));
  _Float16* P_dWhh = (_Float16*)(ws + alloc(256L * 32 * 512 * 2));
  _Float16* P_Wout = (_Float16*)(ws + alloc(5L * 32 * 512 * 2));
  _Float16* P_Wstop = (_Float16*)(ws + alloc(1L * 32 * 512 * 2));
  // activations / shared step state
  _Float16* xemb = (_Float16*)(ws + alloc(200L * 32 * 256 * 2));
  _Float16* decin = (_Float16*)(ws + alloc(400L * 32 * 80 * 2));
  _Float16* enc16 = (_Float16*)(ws + alloc(6400L * 512 * 2));
  float* hf = (float*)(ws + alloc(200L * 32 * 256 * 4));
  float* hb = (float*)(ws + alloc(200L * 32 * 256 * 4));
  float* enc32 = (float*)(ws + alloc(6400L * 512 * 4));
  float* proj = (float*)(ws + alloc(6400L * 1024 * 4));
  _Float16* gxin = (_Float16*)(ws + alloc(32L * 608 * 2));
  _Float16* gh0 = (_Float16*)(ws + alloc(32L * 1024 * 2));
  _Float16* gh1 = (_Float16*)(ws + alloc(32L * 1024 * 2));
  unsigned* bar = (unsigned*)(ws + alloc(256));

  float* out_mel = (float*)d_out;              // (B,T,80)
  float* out_stop = out_mel + 32L * 400 * 80;  // (B,T,1)
  float* out_align = out_stop + 32L * 400;     // (B,T,200)

  // one-time setup
  pack_w<<<512, 256, 0, stream>>>(efWih, 256, 0, 1024, 256, 64, 8, P_efWih);
  pack_w<<<512, 256, 0, stream>>>(efWhh, 256, 0, 1024, 256, 64, 8, P_efWhh);
  pack_w<<<512, 256, 0, stream>>>(ebWih, 256, 0, 1024, 256, 64, 8, P_ebWih);
  pack_w<<<512, 256, 0, stream>>>(ebWhh, 256, 0, 1024, 256, 64, 8, P_ebWhh);
  pack_w<<<1024, 256, 0, stream>>>(Ws, 1536, 0, 1024, 512, 64, 16, P_Wse);
  pack_w<<<2048, 256, 0, stream>>>(Ws, 1536, 512, 1024, 1024, 64, 32, P_Wsd);
  pack_w<<<4096, 256, 0, stream>>>(dWih, 592, 0, 4096, 592, 256, 19, P_dWih);
  pack_w<<<8192, 256, 0, stream>>>(dWhh, 1024, 0, 4096, 1024, 256, 32, P_dWhh);
  pack_w<<<256, 256, 0, stream>>>(Wout, 1024, 0, 80, 1024, 5, 32, P_Wout);
  pack_w<<<32, 256, 0, stream>>>(Wstop, 1024, 0, 1, 1024, 1, 32, P_Wstop);
  embed_k<<<6400, 256, 0, stream>>>(text, emb, xemb);
  decin_k<<<4000, 256, 0, stream>>>(mel, decin);
  zero_bar<<<1, 64, 0, stream>>>(bar);

  // persistent encoder: block 0 = forward, block 1 = backward
  hipFuncSetAttribute((const void*)enc_lstm, hipFuncAttributeMaxDynamicSharedMemorySize,
                      131072 + 32768 + 16384);
  enc_lstm<<<2, 256, 131072 + 32768 + 16384, stream>>>(xemb, P_efWih, P_efWhh, efb,
                                                       P_ebWih, P_ebWhh, ebb, hf, hb);

  assemble_enc<<<12800, 256, 0, stream>>>(hf, hb, enc32, enc16);
  proj_gemm<<<3200, 256, 0, stream>>>(enc16, P_Wse, bs, proj);

  // cooperative 16-WG decoder
  decoder_k<<<NWG, 256, 0, stream>>>(proj, enc32, decin, P_Wsd, P_dWih, P_dWhh, P_Wout,
                                     P_Wstop, v, db, bout, bstop, gxin, gh0, gh1, bar,
                                     out_mel, out_stop, out_align);
}
